// HeadAttention_11974368821408
// MI455X (gfx1250) — matrix operand order repaired and checked by reading, not compiled, not run on hardware
//
#include <hip/hip_runtime.h>
#include <hip/hip_bf16.h>

// B=4, S=4096, E=1024, D=64 single-head causal attention.
// Pipeline:
//   (0) convert W_q/W_k/W_v fp32 -> bf16 once (loop-invariant, L2-resident)
//   (1) fused QKV projection: x (fp32, streamed NT) x Wbf16 -> bf16 Q/K/V, WMMA bf16
//   (2) causal flash attention, WMMA bf16 + TDM (tensor_load_to_lds) V staging

typedef __attribute__((ext_vector_type(16))) __bf16 v16bf;
typedef __attribute__((ext_vector_type(8)))  __bf16 v8bf;
typedef __attribute__((ext_vector_type(8)))  float  v8f;
typedef __attribute__((ext_vector_type(4)))  float  v4f;

typedef __attribute__((ext_vector_type(4))) unsigned int tdm_g0_t;  // D# group 0
typedef __attribute__((ext_vector_type(8))) int          tdm_g1_t;  // D# group 1
typedef __attribute__((ext_vector_type(4))) int          tdm_g4_t;  // D# groups 2/3
typedef __attribute__((ext_vector_type(8))) int          tdm_g8_t;  // trailing group (6-arg form)

#define SLEN 4096
#define EMB  1024
#define HDIM 64

// ---------------------------------------------------------------------------
// Kernel 0: convert the three [64,1024] fp32 weight matrices to bf16.
// wbf layout: [3][64][1024], order Wq, Wk, Wv.
// ---------------------------------------------------------------------------
__global__ __launch_bounds__(256) void wcvt_kernel(
    const float* __restrict__ Wq, const float* __restrict__ Wk, const float* __restrict__ Wv,
    __bf16* __restrict__ wbf)
{
    const int tid = blockIdx.x * 256 + threadIdx.x;      // 49152 threads, 4 floats each
    const int w   = tid / (HDIM * EMB / 4);              // 0..2
    const int e   = tid % (HDIM * EMB / 4);
    const float* W = (w == 0) ? Wq : (w == 1) ? Wk : Wv;
    v4f f = *(const v4f*)(W + (size_t)e * 4);
    __bf16* o = wbf + (size_t)w * HDIM * EMB + (size_t)e * 4;
#pragma unroll
    for (int j = 0; j < 4; ++j) o[j] = (__bf16)f[j];
}

// ---------------------------------------------------------------------------
// Kernel 1: Q/K/V projections.  Each wave: one 16-row tile of x, 12 output
// tiles (3 matrices x 4 n-tiles of 16), K=1024 swept in steps of 32.
// B-fragments come pre-converted from bf16 weights (pure global_load_b128).
// ---------------------------------------------------------------------------
__global__ __launch_bounds__(256) void qkv_proj_kernel(
    const float* __restrict__ x,
    const __bf16* __restrict__ wbf,
    __bf16* __restrict__ qout, __bf16* __restrict__ kout, __bf16* __restrict__ vout)
{
    const int lane  = threadIdx.x & 31;
    const int wave  = threadIdx.x >> 5;
    const int l16   = lane & 15;
    const int half  = lane >> 4;
    const int mtile = blockIdx.x * 8 + wave;   // 1024 row tiles total
    const int m0    = mtile * 16;

    __bf16* Out[3] = { qout, kout, vout };

    v8f zero = {};
    v8f acc[12];
#pragma unroll
    for (int i = 0; i < 12; ++i) acc[i] = zero;

    const int arow = m0 + l16;          // A-matrix row for this lane
    for (int kk = 0; kk < EMB; kk += 32) {
        // A fragment (16x32 bf16): lane<16 -> K {0..7,16..23}, lane>=16 -> {8..15,24..31}
        const float* xp = x + (size_t)arow * EMB + kk + half * 8;
        v4f x0 = __builtin_nontemporal_load((const v4f*)(xp));        // streamed once
        v4f x1 = __builtin_nontemporal_load((const v4f*)(xp + 4));
        v4f x2 = __builtin_nontemporal_load((const v4f*)(xp + 16));
        v4f x3 = __builtin_nontemporal_load((const v4f*)(xp + 20));
        v16bf a;
#pragma unroll
        for (int j = 0; j < 4; ++j) {
            a[j]      = (__bf16)x0[j];
            a[4 + j]  = (__bf16)x1[j];
            a[8 + j]  = (__bf16)x2[j];
            a[12 + j] = (__bf16)x3[j];
        }
        if (kk + 32 < EMB) __builtin_prefetch(xp + 32, 0, 1);   // global_prefetch_b8

#pragma unroll
        for (int w = 0; w < 3; ++w) {
#pragma unroll
            for (int nt = 0; nt < 4; ++nt) {
                // B = W^T chunk: lane column n = nt*16+l16, K {0..7,16..23} / {8..15,24..31} by lane-half
                const __bf16* wp = wbf + ((size_t)w * HDIM + nt * 16 + l16) * EMB
                                       + kk + half * 8;
                v8bf wb0 = *(const v8bf*)wp;
                v8bf wb1 = *(const v8bf*)(wp + 16);
                v16bf b;
#pragma unroll
                for (int j = 0; j < 8; ++j) { b[j] = wb0[j]; b[8 + j] = wb1[j]; }
                acc[w * 4 + nt] = __builtin_amdgcn_wmma_f32_16x16x32_bf16(
                    false, a, false, b, (short)0, acc[w * 4 + nt], false, false);
            }
        }
    }

    // C layout: vgpr r, lanes 0-15 -> row r, lanes 16-31 -> row r+8; col = lane&15
#pragma unroll
    for (int w = 0; w < 3; ++w)
#pragma unroll
        for (int nt = 0; nt < 4; ++nt)
#pragma unroll
            for (int r = 0; r < 8; ++r) {
                int m = m0 + r + 8 * half;
                int c = nt * 16 + l16;
                Out[w][(size_t)m * HDIM + c] = (__bf16)acc[w * 4 + nt][r];
            }
}

// ---------------------------------------------------------------------------
// Kernel 2: causal flash attention.  Each wave: 16 query rows, kv chunks of 32.
// Per chunk: TDM-stages the V tile into LDS (TENSORcnt) overlapped with
// 4 WMMA (Q.K^T) + online softmax, then 4 WMMA (P.V).
// ---------------------------------------------------------------------------
__global__ __launch_bounds__(256) void flash_attn_kernel(
    const __bf16* __restrict__ Q, const __bf16* __restrict__ K, const __bf16* __restrict__ V,
    float* __restrict__ out)
{
    __shared__ __bf16 v_lds[8][32 * HDIM];   // per wave: V tile row-major [k][d] 32x64
    __shared__ __bf16 p_lds[8][16 * 32];     // per wave: P tile [m][n] 16x32

    const int lane = threadIdx.x & 31;
    const int wave = threadIdx.x >> 5;
    const int l16  = lane & 15;
    const int half = lane >> 4;
    const int b    = blockIdx.y;
    const int q0   = blockIdx.x * 128 + wave * 16;

    const __bf16* Qb = Q + (size_t)b * SLEN * HDIM;
    const __bf16* Kb = K + (size_t)b * SLEN * HDIM;
    const __bf16* Vb = V + (size_t)b * SLEN * HDIM;
    __bf16* vt = v_lds[wave];
    __bf16* pt = p_lds[wave];

    // Loop-invariant part of the TDM descriptor (ISA 08_async_tensor §8).
    // Group1: data_size=2B (code 1), tensor_dim0=64, tensor_dim1=32,
    //         tile_dim0=64, tile_dim1=32, tensor_dim0_stride=64. 2D tensor.
    const unsigned lds_base = (unsigned)(size_t)vt;
    const tdm_g1_t g1 = { 0x00010000,          // workgroup_mask=0, data_size=1(2B)
                          (64 << 16),          // tensor_dim0 low16 (=64)
                          (32 << 16),          // tensor_dim0 hi=0 | tensor_dim1 low16 (=32)
                          (64 << 16),          // tensor_dim1 hi=0 | tile_dim0 (=64)
                          32,                  // tile_dim1=32, tile_dim2=0
                          64,                  // tensor_dim0_stride low32 (=64)
                          0, 0 };              // stride hi, tensor_dim1_stride
    const tdm_g4_t gz4 = { 0, 0, 0, 0 };       // 2D: groups 2/3 unused
    const tdm_g8_t gz8 = { 0, 0, 0, 0, 0, 0, 0, 0 };

    // Q A-fragments for d-halves 0..31 / 32..63 (loaded once)
    v16bf aq[2];
#pragma unroll
    for (int dh = 0; dh < 2; ++dh) {
        const __bf16* qp = Qb + (size_t)(q0 + l16) * HDIM + dh * 32 + half * 8;
        v8bf c0 = *(const v8bf*)qp;
        v8bf c1 = *(const v8bf*)(qp + 16);
#pragma unroll
        for (int j = 0; j < 8; ++j) { aq[dh][j] = c0[j]; aq[dh][8 + j] = c1[j]; }
    }

    v8f zero = {};
    float mrow[8], lrow[8];
    v8f o[4];
#pragma unroll
    for (int r = 0; r < 8; ++r) { mrow[r] = -3.0e38f; lrow[r] = 0.0f; }
#pragma unroll
    for (int t = 0; t < 4; ++t) o[t] = zero;

    const int qmax = q0 + 15;
    for (int kv0 = 0; kv0 <= qmax; kv0 += 32) {
        // ---- kick off TDM: V[kv0..kv0+31][0..63] -> LDS (row-major), TENSORcnt ----
        {
            unsigned long long ga = (unsigned long long)(size_t)(Vb + (size_t)kv0 * HDIM);
            tdm_g0_t g0 = { 1u,                               // count=1, user mode
                            lds_base,                         // lds_addr
                            (unsigned)(ga & 0xffffffffu),     // global_addr[31:0]
                            (unsigned)(ga >> 32) | 0x80000000u }; // addr[56:32] | type=2
            __builtin_amdgcn_tensor_load_to_lds(g0, g1, gz4, gz4, gz8, 0);
        }

        // ---- scores S = Q (16x64) . K^T (64x32), 4 WMMA (overlaps the DMA) ----
        v8f s[2] = { zero, zero };
#pragma unroll
        for (int nt = 0; nt < 2; ++nt)
#pragma unroll
            for (int dh = 0; dh < 2; ++dh) {
                // B = K^T frag: lane column = kv0+nt*16+l16; K-dim {0..7,16..23} / {8..15,24..31} by lane-half
                const __bf16* kp = Kb + (size_t)(kv0 + nt * 16 + l16) * HDIM + dh * 32 + half * 8;
                v8bf kc0 = *(const v8bf*)kp;
                v8bf kc1 = *(const v8bf*)(kp + 16);
                v16bf bk;
#pragma unroll
                for (int j = 0; j < 8; ++j) { bk[j] = kc0[j]; bk[8 + j] = kc1[j]; }
                s[nt] = __builtin_amdgcn_wmma_f32_16x16x32_bf16(
                    false, aq[dh], false, bk, (short)0, s[nt], false, false);
            }

        // ---- scale + causal mask + online softmax (row = r + 8*half) ----
        float alpha[8];
#pragma unroll
        for (int r = 0; r < 8; ++r) {
            int qrow = q0 + r + 8 * half;
#pragma unroll
            for (int nt = 0; nt < 2; ++nt) {
                int kvpos = kv0 + nt * 16 + l16;
                float v = s[nt][r] * 0.125f;             // 1/sqrt(64)
                s[nt][r] = (kvpos <= qrow) ? v : -3.0e38f;
            }
            float mx = fmaxf(s[0][r], s[1][r]);
            mx = fmaxf(mx, __shfl_xor(mx, 1, 32));
            mx = fmaxf(mx, __shfl_xor(mx, 2, 32));
            mx = fmaxf(mx, __shfl_xor(mx, 4, 32));
            mx = fmaxf(mx, __shfl_xor(mx, 8, 32));
            float mnew = fmaxf(mrow[r], mx);
            alpha[r] = __expf(mrow[r] - mnew);
            mrow[r] = mnew;
            float rs = 0.0f;
#pragma unroll
            for (int nt = 0; nt < 2; ++nt) {
                float p = __expf(s[nt][r] - mnew);
                s[nt][r] = p;
                rs += p;
            }
            rs += __shfl_xor(rs, 1, 32);
            rs += __shfl_xor(rs, 2, 32);
            rs += __shfl_xor(rs, 4, 32);
            rs += __shfl_xor(rs, 8, 32);
            lrow[r] = alpha[r] * lrow[r] + rs;
        }

        // ---- P (C layout) -> LDS bf16 row-major 16x32 (transpose to A layout) ----
#pragma unroll
        for (int nt = 0; nt < 2; ++nt)
#pragma unroll
            for (int r = 0; r < 8; ++r)
                pt[(r + 8 * half) * 32 + nt * 16 + l16] = (__bf16)s[nt][r];

        // ---- rescale running O ----
#pragma unroll
        for (int t = 0; t < 4; ++t)
#pragma unroll
            for (int r = 0; r < 8; ++r) o[t][r] *= alpha[r];

        // ---- P A-fragment from LDS ----
        v16bf ap;
        {
            const __bf16* pp = pt + l16 * 32 + half * 8;
            v8bf c0 = *(const v8bf*)pp;
            v8bf c1 = *(const v8bf*)(pp + 16);
#pragma unroll
            for (int j = 0; j < 8; ++j) { ap[j] = c0[j]; ap[8 + j] = c1[j]; }
        }

        // ---- wait for the TDM tile, then O += P (16x32) . V (32x64), 4 WMMA ----
        __builtin_amdgcn_s_wait_tensorcnt(0);
#pragma unroll
        for (int t = 0; t < 4; ++t) {
            // B frag from row-major LDS V: lane column d = t*16+l16, K = half*8 + (j&7) + (j>>3)*16
            v16bf bv;
#pragma unroll
            for (int j = 0; j < 16; ++j)
                bv[j] = vt[(half * 8 + (j & 7) + (j >> 3) * 16) * HDIM + t * 16 + l16];
            o[t] = __builtin_amdgcn_wmma_f32_16x16x32_bf16(
                false, ap, false, bv, (short)0, o[t], false, false);
        }
    }

    // ---- epilogue: normalize by row sum, store fp32 [B,S,64] (written once -> NT) ----
#pragma unroll
    for (int t = 0; t < 4; ++t)
#pragma unroll
        for (int r = 0; r < 8; ++r) {
            int m = q0 + r + 8 * half;
            int c = t * 16 + l16;
            __builtin_nontemporal_store(o[t][r] / lrow[r],
                                        out + ((size_t)b * SLEN + m) * HDIM + c);
        }
}

// ---------------------------------------------------------------------------
extern "C" void kernel_launch(void* const* d_in, const int* in_sizes, int n_in,
                              void* d_out, int out_size, void* d_ws, size_t ws_size,
                              hipStream_t stream) {
    (void)in_sizes; (void)n_in; (void)out_size; (void)ws_size;
    const float* x  = (const float*)d_in[0];
    const float* Wk = (const float*)d_in[1];
    const float* Wq = (const float*)d_in[2];
    const float* Wv = (const float*)d_in[3];
    float* out = (float*)d_out;

    const size_t qkv_elems = (size_t)4 * SLEN * HDIM;   // 1M elems each
    __bf16* q   = (__bf16*)d_ws;
    __bf16* k   = q + qkv_elems;
    __bf16* v   = k + qkv_elems;
    __bf16* wbf = v + qkv_elems;                        // [3][64][1024] bf16

    // Convert weights once: 3*64*1024 = 196608 elems, 4 per thread.
    wcvt_kernel<<<dim3(192), dim3(256), 0, stream>>>(Wq, Wk, Wv, wbf);
    // 1024 row-tiles of 16 / 8 waves per block = 128 blocks
    qkv_proj_kernel<<<dim3(128), dim3(256), 0, stream>>>(x, wbf, q, k, v);
    // 32 q-blocks of 128 rows x 4 batches
    flash_attn_kernel<<<dim3(32, 4), dim3(256), 0, stream>>>(q, k, v, out);
}
